// ParallelBackwardRnn_76510547411001
// MI455X (gfx1250) — matrix operand order repaired and checked by reading, not compiled, not run on hardware
//
#include <hip/hip_runtime.h>
#include <hip/hip_bf16.h>

// ---------------- CDNA5 WMMA types ----------------
typedef __attribute__((ext_vector_type(16))) __bf16 v16bf;
typedef __attribute__((ext_vector_type(8)))  float  v8f;
typedef __attribute__((ext_vector_type(4)))  float  f32x4;   // native vec for NT loads
typedef __attribute__((ext_vector_type(2)))  float  f32x2;
typedef __attribute__((ext_vector_type(2)))  __bf16 bf16x2;

union ABFrag {
    v16bf v;
    uint4 q[2];           // two 16B chunks = 32B = 8 VGPRs
};

// packed f32x2 -> bf16x2 (native v_cvt_pk_bf16_f32 path, RNE)
__device__ __forceinline__ unsigned pack2bf(float a, float b) {
    union { bf16x2 h; unsigned u; } c;
    c.h = __builtin_convertvector((f32x2){a, b}, bf16x2);
    return c.u;
}
// scalar f32 -> bf16 bits (native cvt, RNE)
__device__ __forceinline__ unsigned short f2bf(float f) {
    union { __bf16 h; unsigned short u; } c;
    c.h = (__bf16)f;
    return c.u;
}

#define B_  64
#define S_  512
#define D_  1024
#define H_  1024

// ==================================================================
// Phase 1: xw = x @ Wx + b        (M=32768, N=1024, K=1024)
// block tile 128x128, 256 threads = 8 waves in 2(M) x 4(N) layout,
// each wave owns a 64x32 region = 4x2 wmma accumulators.
// x loads are non-temporal (streamed once) so the 134MB xw output can
// stay resident in the 192MB L2 for phase 2; next K-chunk prefetched.
// ==================================================================
#define G_TM 128
#define G_TN 128
#define G_TK 32
#define G_LDA 40   // padded LDS stride (bf16 elems)
#define G_LDB 40

__global__ __launch_bounds__(256)
void rnn_xw_gemm(const float* __restrict__ x,    // [M][K]
                 const float* __restrict__ Wx,   // [K][N]
                 const float* __restrict__ bias, // [N]
                 float* __restrict__ out,        // [M][N]
                 int M, int N, int K)
{
    __shared__ unsigned short As[G_TM * G_LDA];  // row-major  [m][k]
    __shared__ unsigned short Bs[G_TN * G_LDB];  // col-major  [n][k]

    const int tid  = threadIdx.x;
    const int lane = tid & 31;
    const int wave = tid >> 5;
    const int wm   = wave >> 2;          // 0..1
    const int wn   = wave & 3;           // 0..3
    const int mBase = blockIdx.y * G_TM;
    const int nBase = blockIdx.x * G_TN;

    v8f acc[4][2];
    v8f zero = {};
#pragma unroll
    for (int i = 0; i < 4; ++i)
#pragma unroll
        for (int j = 0; j < 2; ++j) acc[i][j] = zero;

    const int l16 = lane & 15;
    const int kbA = (lane >> 4) * 8;     // A: K-base per half-wave
    const int kbB = (lane >> 4) * 8;     // B: K-base per half-wave

    // staging geometry (constant per thread)
    const int a_kc = (tid & 7) * 4;
    const int a_r0 = tid >> 3;                       // 0..31
    const int b_nc = (tid & 31) * 4;
    const int b_k0 = tid >> 5;                       // 0..7
    const float* xRow  = x  + (size_t)(mBase + a_r0) * K + a_kc;   // + p*32*K + k0
    const float* wxRow = Wx + (size_t)b_k0 * N + nBase + b_nc;     // + p*8*N  + k0*N

    for (int k0 = 0; k0 < K; k0 += G_TK) {
        // ---- issue all global loads first (one clause, one wait) ----
        f32x4 fa[4], fb[4];
#pragma unroll
        for (int p = 0; p < 4; ++p)
            fa[p] = __builtin_nontemporal_load(
                (const f32x4*)(xRow + (size_t)p * 32 * K + k0));
#pragma unroll
        for (int p = 0; p < 4; ++p)
            fb[p] = *(const f32x4*)(wxRow + (size_t)(k0 + p * 8) * N);
        // prefetch next K-chunk of x (global_prefetch_b8)
        if (k0 + G_TK < K)
            __builtin_prefetch(xRow + k0 + G_TK, 0, 1);

        __syncthreads();
        // ---- stage A tile: 128x32 bf16 LDS (row major) ----
#pragma unroll
        for (int p = 0; p < 4; ++p) {
            unsigned* dst = (unsigned*)&As[(a_r0 + p * 32) * G_LDA + a_kc];
            dst[0] = pack2bf(fa[p].x, fa[p].y);
            dst[1] = pack2bf(fa[p].z, fa[p].w);
        }
        // ---- stage B tile: 32x128 bf16 LDS (col major) ----
#pragma unroll
        for (int p = 0; p < 4; ++p) {
            const int kk = b_k0 + p * 8;
            Bs[(b_nc + 0) * G_LDB + kk] = f2bf(fb[p].x);
            Bs[(b_nc + 1) * G_LDB + kk] = f2bf(fb[p].y);
            Bs[(b_nc + 2) * G_LDB + kk] = f2bf(fb[p].z);
            Bs[(b_nc + 3) * G_LDB + kk] = f2bf(fb[p].w);
        }
        __syncthreads();

        // ---- fragments + WMMA ----
        ABFrag a[4], b[2];
#pragma unroll
        for (int mi = 0; mi < 4; ++mi) {
            const unsigned short* ap = &As[(wm * 64 + mi * 16 + l16) * G_LDA + kbA];
            a[mi].q[0] = *(const uint4*)(ap);
            a[mi].q[1] = *(const uint4*)(ap + 16);
        }
#pragma unroll
        for (int ni = 0; ni < 2; ++ni) {
            const unsigned short* bp = &Bs[(wn * 32 + ni * 16 + l16) * G_LDB + kbB];
            b[ni].q[0] = *(const uint4*)(bp);
            b[ni].q[1] = *(const uint4*)(bp + 16);
        }
#pragma unroll
        for (int mi = 0; mi < 4; ++mi)
#pragma unroll
            for (int ni = 0; ni < 2; ++ni)
                acc[mi][ni] = __builtin_amdgcn_wmma_f32_16x16x32_bf16(
                    false, a[mi].v, false, b[ni].v, (short)0, acc[mi][ni], false, false);
    }

    // ---- epilogue: + bias, store f32 (RT: keep xw hot in L2) ----
    const int rhi = (lane >> 4) * 8;
#pragma unroll
    for (int ni = 0; ni < 2; ++ni) {
        const int col = nBase + wn * 32 + ni * 16 + l16;
        const float bv = bias[col];
        float* op = out + (size_t)(mBase + wm * 64 + rhi) * N + col;
#pragma unroll
        for (int mi = 0; mi < 4; ++mi)
#pragma unroll
            for (int r = 0; r < 8; ++r)
                op[(size_t)(mi * 16 + r) * N] = acc[mi][ni][r] + bv;
    }
}

// ==================================================================
// Phase 2: persistent recurrent kernel.
// grid = 16 col-tiles x 4 batch-tiles = 64 blocks, 128 threads (4 waves).
// Each block keeps its 64-column Wh slice in LDS (bf16, col-major) for
// all 512 steps. Per step: each wave does one 16x16 output tile with
// 32 x v_wmma_f32_16x16x32_bf16, then tanh, then device-wide barrier.
// All global addresses are strength-reduced to pointer increments so
// the per-step critical path has no 64-bit multiplies.
// ==================================================================
#define PK 1032   // padded K stride in LDS (bf16 elems): 2064B = 516 dw -> 4-bank skew

__global__ __launch_bounds__(128)
void rnn_steps(const float* __restrict__ Wh,         // [H][H] f32
               float* __restrict__ out,              // [B][S][H]; holds xw on entry
               unsigned short* __restrict__ hbuf,    // [2][B][H] bf16 (buf0 zeroed)
               int* __restrict__ bar)
{
    extern __shared__ unsigned short Ws[];           // [64][PK] bf16, col-major

    const int tid  = threadIdx.x;
    const int lane = tid & 31;
    const int wave = tid >> 5;                       // 0..3
    const int colBase = blockIdx.x * 64;
    const int rowBase = blockIdx.y * 16;
    const int nBlocks = (int)(gridDim.x * gridDim.y);

    // ---- stage Wh slice once: f32 -> bf16, col-major into LDS ----
    // each thread handles 4 consecutive columns per k row (vector loads)
    for (int base = tid * 4; base < 64 * H_; base += 128 * 4) {
        const int c = base & 63;
        const int k = base >> 6;
        const f32x4 f = *(const f32x4*)(Wh + (size_t)k * H_ + colBase + c);
        Ws[(c + 0) * PK + k] = f2bf(f.x);
        Ws[(c + 1) * PK + k] = f2bf(f.y);
        Ws[(c + 2) * PK + k] = f2bf(f.z);
        Ws[(c + 3) * PK + k] = f2bf(f.w);
    }
    __syncthreads();

    const int l16 = lane & 15;
    const int kbA = (lane >> 4) * 8;
    const int kbB = (lane >> 4) * 8;
    const int rhi = (lane >> 4) * 8;
    const int col = colBase + wave * 16 + l16;
    const unsigned short* wsBase = &Ws[(wave * 16 + l16) * PK];

    // ---- precomputed global pointers (strength reduction) ----
    // xw/h output: one pointer per accumulator row, advanced by H_ per step
    float* op[8];
#pragma unroll
    for (int r = 0; r < 8; ++r)
        op[r] = out + (size_t)(rowBase + rhi + r) * S_ * H_ + col;
    // h-state: ping-pong bases; row offsets r*H_*2B fit the 24-bit imm field
    unsigned short* hnBase = hbuf + (size_t)(rowBase + rhi) * H_ + col;
    const unsigned short* hpBase = hbuf + (size_t)(rowBase + l16) * H_;

    for (int t = 0; t < S_; ++t) {
        const int pPrev = (t & 1) * (B_ * H_);
        const int pNext = ((t + 1) & 1) * (B_ * H_);

        // C init = xw (stored in out[:, t, :])
        v8f acc;
#pragma unroll
        for (int r = 0; r < 8; ++r)
            acc[r] = *op[r];

        // acc += h_{t-1} @ Wh   (K = 1024 in chunks of 32)
        const unsigned short* aRow = hpBase + pPrev;
#pragma unroll 4
        for (int k0 = 0; k0 < H_; k0 += 32) {
            ABFrag a, b;
            a.q[0] = *(const uint4*)(aRow + k0 + kbA);
            a.q[1] = *(const uint4*)(aRow + k0 + kbA + 16);
            b.q[0] = *(const uint4*)(wsBase + k0 + kbB);
            b.q[1] = *(const uint4*)(wsBase + k0 + kbB + 16);
            acc = __builtin_amdgcn_wmma_f32_16x16x32_bf16(
                false, a.v, false, b.v, (short)0, acc, false, false);
        }

        // h_t = tanh(acc): overwrite xw slot, publish bf16 state
        unsigned short* hn = hnBase + pNext;
#pragma unroll
        for (int r = 0; r < 8; ++r) {
            const float h = tanhf(acc[r]);
            *op[r] = h;
            hn[r * H_] = f2bf(h);
            op[r] += H_;                 // next timestep slot
        }

        // ---- device-wide barrier (release/acquire) ----
        __threadfence();
        __syncthreads();
        if (tid == 0) {
            atomicAdd(bar, 1);
            const int target = nBlocks * (t + 1);
            while (*(volatile int*)bar < target)
                __builtin_amdgcn_s_sleep(2);
        }
        __syncthreads();
        __threadfence();
    }
}

// ==================================================================
extern "C" void kernel_launch(void* const* d_in, const int* in_sizes, int n_in,
                              void* d_out, int out_size, void* d_ws, size_t ws_size,
                              hipStream_t stream) {
    (void)in_sizes; (void)n_in; (void)out_size; (void)ws_size;
    const float* x  = (const float*)d_in[0];   // [B][S][D]
    const float* Wx = (const float*)d_in[1];   // [D][H]
    const float* Wh = (const float*)d_in[2];   // [H][H]
    const float* b  = (const float*)d_in[3];   // [H]
    float* out = (float*)d_out;                // [B][S][H]

    int* bar = (int*)d_ws;                                        // barrier counter
    unsigned short* hbuf = (unsigned short*)((char*)d_ws + 256);  // [2][B][H] bf16

    // zero barrier + h0 every launch (deterministic across graph replays)
    (void)hipMemsetAsync(d_ws, 0, 256 + (size_t)2 * B_ * H_ * sizeof(unsigned short), stream);

    // Phase 1: xw = x @ Wx + b  -> out
    dim3 g1(H_ / G_TN, (B_ * S_) / G_TM);   // (8, 256)
    rnn_xw_gemm<<<g1, 256, 0, stream>>>(x, Wx, b, out, B_ * S_, H_, D_);

    // Phase 2: 512 recurrent steps, persistent 64-block grid
    dim3 g2(H_ / 64, B_ / 16);              // (16, 4)
    const size_t ldsBytes = (size_t)64 * PK * sizeof(unsigned short); // ~129 KB
    rnn_steps<<<g2, 128, ldsBytes, stream>>>(Wh, out, hbuf, bar);
}
